// TDspeMamba_55937654063132
// MI455X (gfx1250) — compile-verified, matrix operand order checked by reading, not run on hardware
//
#include <hip/hip_runtime.h>

typedef __attribute__((ext_vector_type(16))) _Float16 v16h;
typedef __attribute__((ext_vector_type(8)))  float    v8f;

#define BB 4
#define CC 128
#define HWP 16384              // H*W
#define NELEM (BB*CC*HWP)      // 8388608 per tensor
#define NTOK (BB*HWP)          // 65536 sequences
#define DIN 256
#define DBCW 48                // 40 padded to 48
#define XSTR 136               // x_lds stride in halves (128+8)
#define XCSTR 264              // xc/s stride in halves (256+8)
#define OSTR 132               // out stride in f32 (128+4)

#define X_OFF    0
#define XC_OFF   (X_OFF + 32*XSTR*2)          // 8704
#define S_OFF    (XC_OFF + 32*XCSTR*2)        // 25600
#define DBC_OFF  (S_OFF + 32*XCSTR*2)         // 42496 (S region reused as f32 out: 32*132*4 B)
#define NEGA_OFF (DBC_OFF + 32*DBCW*4)        // 48640
#define SMEM_BYTES (NEGA_OFF + 256*16*4)      // 65024

// f16 weight pack layout in d_ws (halves)
#define WIN_HALFS (512*128)
#define XP_ROWS 48
#define XP_HALFS (XP_ROWS*256)
#define OP_HALFS (128*256)
#define WS_WIN 0
#define WS_XP  (WIN_HALFS)
#define WS_OP  (WIN_HALFS + XP_HALFS)
#define WS_TOTAL_HALFS (WIN_HALFS + XP_HALFS + OP_HALFS)   // 110592

__device__ __forceinline__ float silu_f(float x) { return x / (1.f + __expf(-x)); }
__device__ __forceinline__ float softplus_f(float x) {
  return (x > 20.f) ? x : __logf(1.f + __expf(x));
}
__device__ __forceinline__ v8f wmma16(v16h a, v16h b, v8f c) {
  return __builtin_amdgcn_wmma_f32_16x16x32_f16(false, a, false, b, (short)0, c, false, false);
}
// A fragment (16 rows x 32 K, f16) from LDS row-major tile.
// Lane L: row M = L&15; lanes 0-15 hold K 0-7 / 16-23, lanes 16-31 hold K 8-15 / 24-31.
__device__ __forceinline__ v16h lds_load_A(const _Float16* base, int stride, int lane) {
  const int m = lane & 15, half = lane >> 4;
  const _Float16* p = base + m * stride + half * 8;
  v16h a;
#pragma unroll
  for (int j = 0; j < 8; ++j) a[j] = p[j];
#pragma unroll
  for (int j = 0; j < 8; ++j) a[8 + j] = p[16 + j];
  return a;
}
// B fragment (32 K x 16 N) from prepacked row-major f16 weight; lane holds column n.
// Two aligned 16-byte loads per fragment, no converts, no bounds branch.
__device__ __forceinline__ v16h load_B_f16(const _Float16* __restrict__ W, int ldk,
                                           int row0, int k0, int lane) {
  const int n = lane & 15, half = lane >> 4;
  const _Float16* p = W + (row0 + n) * ldk + k0 + half * 8;
  v16h b;
#pragma unroll
  for (int j = 0; j < 8; ++j) b[j] = p[j];
#pragma unroll
  for (int j = 0; j < 8; ++j) b[8 + j] = p[16 + j];
  return b;
}

// ---- Prep: convert weights to f16 (x_proj padded 40->48 rows with zeros) ----
__global__ __launch_bounds__(256) void weights_prep_kernel(
    const float* __restrict__ in_proj_w, const float* __restrict__ x_proj_w,
    const float* __restrict__ out_proj_w, _Float16* __restrict__ ws) {
  const int i = blockIdx.x * 256 + threadIdx.x;
  if (i < WIN_HALFS) {
    ws[WS_WIN + i] = (_Float16)in_proj_w[i];
  } else if (i < WS_XP + XP_HALFS) {
    const int j = i - WS_XP;
    const int row = j >> 8;                       // /256
    ws[i] = (_Float16)(row < 40 ? x_proj_w[j] : 0.f);
  } else if (i < WS_TOTAL_HALFS) {
    ws[i] = (_Float16)out_proj_w[i - WS_OP];
  }
}

__global__ __launch_bounds__(32) void mamba2d_fused_kernel(
    const float* __restrict__ x1, const float* __restrict__ x2,
    const float* __restrict__ conv_w, const float* __restrict__ conv_b,
    const float* __restrict__ dt_proj_w, const float* __restrict__ dt_proj_b,
    const float* __restrict__ A_log, const float* __restrict__ Dvec,
    const float* __restrict__ ln_w, const float* __restrict__ ln_b,
    const _Float16* __restrict__ wf16, float* __restrict__ out) {
  __shared__ __align__(16) char smem[SMEM_BYTES];
  _Float16* x_lds   = (_Float16*)(smem + X_OFF);    // [step*16+pos][XSTR] input f16
  _Float16* xc_lds  = (_Float16*)(smem + XC_OFF);   // xc, later ypre
  _Float16* s_lds   = (_Float16*)(smem + S_OFF);    // silu(z)
  float*    o_lds   = (float*)(smem + S_OFF);       // reuse: out_proj result f32
  float*    dbc_lds = (float*)(smem + DBC_OFF);     // [step*16+pos][48]
  float*    negA    = (float*)(smem + NEGA_OFF);    // [256][16]  -exp(A_log)

  const _Float16* Win = wf16 + WS_WIN;              // 512 x 128
  const _Float16* Xp  = wf16 + WS_XP;               // 48  x 256 (zero padded)
  const _Float16* Op  = wf16 + WS_OP;               // 128 x 256

  const int lane = threadIdx.x;      // one wave per block
  const int p0 = blockIdx.x * 16;

  // ---------- Phase 0: stage inputs (f16) + precompute -exp(A_log) ----------
  {
    const int pos = lane & 15, chalf = lane >> 4;
    const int p = p0 + pos, b = p >> 14, hw = p & (HWP - 1);
    const float* xg0 = x1 + (size_t)b * CC * HWP + hw;
    const float* xg1 = x2 + (size_t)b * CC * HWP + hw;
#pragma unroll 4
    for (int c2 = 0; c2 < 64; ++c2) {
      const int c = chalf * 64 + c2;
      x_lds[pos * XSTR + c]        = (_Float16)xg0[c * HWP];
      x_lds[(16 + pos) * XSTR + c] = (_Float16)xg1[c * HWP];
    }
#pragma unroll 4
    for (int i = 0; i < 128; ++i) {
      const int idx = lane * 128 + i;
      negA[idx] = -__expf(A_log[idx]);
    }
    __builtin_prefetch(Win, 0, 0);
    __builtin_prefetch(Xp, 0, 0);
    __builtin_prefetch(Op, 0, 0);
  }
  __syncthreads();

  // ---------- Phase 1: in_proj (WMMA) + conv/silu + dbc accumulation ----------
  v8f dbc[2][3];
#pragma unroll
  for (int s = 0; s < 2; ++s)
#pragma unroll
    for (int t = 0; t < 3; ++t)
#pragma unroll
      for (int r = 0; r < 8; ++r) dbc[s][t][r] = 0.f;

#pragma unroll 1
  for (int dc = 0; dc < 8; ++dc) {
    const int d0 = dc * 32;
    v8f ax[2][2], az[2][2];
#pragma unroll
    for (int s = 0; s < 2; ++s)
#pragma unroll
      for (int u = 0; u < 2; ++u)
#pragma unroll
        for (int r = 0; r < 8; ++r) { ax[s][u][r] = 0.f; az[s][u][r] = 0.f; }

#pragma unroll
    for (int kk = 0; kk < 4; ++kk) {
      const v16h a0 = lds_load_A(x_lds + kk * 32, XSTR, lane);
      const v16h a1 = lds_load_A(x_lds + 16 * XSTR + kk * 32, XSTR, lane);
#pragma unroll
      for (int u = 0; u < 2; ++u) {
        const v16h bx = load_B_f16(Win, CC, d0 + u * 16, kk * 32, lane);
        const v16h bz = load_B_f16(Win, CC, DIN + d0 + u * 16, kk * 32, lane);
        ax[0][u] = wmma16(a0, bx, ax[0][u]);
        ax[1][u] = wmma16(a1, bx, ax[1][u]);
        az[0][u] = wmma16(a0, bz, az[0][u]);
        az[1][u] = wmma16(a1, bz, az[1][u]);
      }
    }
    // conv over L=2 (kw=4, causal pad): xc0 = silu(x0*w3+b), xc1 = silu(x0*w2+x1*w3+b)
#pragma unroll
    for (int u = 0; u < 2; ++u) {
      const int dcol = d0 + u * 16 + (lane & 15);
      const float w2 = conv_w[dcol * 4 + 2], w3 = conv_w[dcol * 4 + 3], cb = conv_b[dcol];
      const int mbase = (lane >> 4) * 8;
#pragma unroll
      for (int r = 0; r < 8; ++r) {
        const int m = mbase + r;
        const float xa = ax[0][u][r], xb = ax[1][u][r];
        xc_lds[m * XCSTR + dcol]        = (_Float16)silu_f(xa * w3 + cb);
        xc_lds[(16 + m) * XCSTR + dcol] = (_Float16)silu_f(xa * w2 + xb * w3 + cb);
        s_lds[m * XCSTR + dcol]         = (_Float16)silu_f(az[0][u][r]);
        s_lds[(16 + m) * XCSTR + dcol]  = (_Float16)silu_f(az[1][u][r]);
      }
    }
    // dbc += xc_chunk @ x_proj_w.T  (40 cols padded to 48, zero rows in pack)
#pragma unroll
    for (int s = 0; s < 2; ++s) {
      const v16h a = lds_load_A(xc_lds + (s * 16) * XCSTR + d0, XCSTR, lane);
#pragma unroll
      for (int t = 0; t < 3; ++t) {
        const v16h b = load_B_f16(Xp, DIN, t * 16, d0, lane);
        dbc[s][t] = wmma16(a, b, dbc[s][t]);
      }
    }
  }
  // spill dbc (dt 0..7 | B 8..23 | C 24..39)
  {
    const int mbase = (lane >> 4) * 8, n = lane & 15;
#pragma unroll
    for (int s = 0; s < 2; ++s)
#pragma unroll
      for (int t = 0; t < 3; ++t)
#pragma unroll
        for (int r = 0; r < 8; ++r)
          dbc_lds[(s * 16 + mbase + r) * DBCW + t * 16 + n] = dbc[s][t][r];
  }
  __syncthreads();

  // ---------- Phase 2: dt_proj + softplus + 2-step selective scan (per lane) ----------
  {
    const int pos = lane & 15, dh = lane >> 4;
    const float* rb0 = dbc_lds + pos * DBCW;
    const float* rb1 = dbc_lds + (16 + pos) * DBCW;
    float dtp0[8], dtp1[8];
#pragma unroll
    for (int r = 0; r < 8; ++r) { dtp0[r] = rb0[r]; dtp1[r] = rb1[r]; }
    float d00 = 0.f, d11 = 0.f, b0c1[16];
#pragma unroll
    for (int s = 0; s < 16; ++s) {
      const float b0 = rb0[8 + s], c0 = rb0[24 + s];
      const float b1 = rb1[8 + s], c1 = rb1[24 + s];
      d00 += b0 * c0; d11 += b1 * c1; b0c1[s] = b0 * c1;
    }
#pragma unroll 1
    for (int dd = 0; dd < 128; ++dd) {
      const int d = dh * 128 + dd;
      float t0 = dt_proj_b[d], t1 = t0;
#pragma unroll
      for (int r = 0; r < 8; ++r) {
        const float w = dt_proj_w[d * 8 + r];
        t0 += dtp0[r] * w; t1 += dtp1[r] * w;
      }
      t0 = softplus_f(t0); t1 = softplus_f(t1);
      const float xc0 = (float)xc_lds[pos * XCSTR + d];
      const float xc1 = (float)xc_lds[(16 + pos) * XCSTR + d];
      float acc = 0.f;
      const float* nA = negA + d * 16;
#pragma unroll
      for (int s = 0; s < 16; ++s) acc += __expf(t1 * nA[s]) * b0c1[s];
      const float Dd = Dvec[d];
      const float g0 = t0 * xc0;
      const float y0 = (g0 * d00 + xc0 * Dd) * (float)s_lds[pos * XCSTR + d];
      const float y1 = (t1 * xc1 * d11 + g0 * acc + xc1 * Dd) * (float)s_lds[(16 + pos) * XCSTR + d];
      xc_lds[pos * XCSTR + d]        = (_Float16)y0;   // ypre overwrites xc in place
      xc_lds[(16 + pos) * XCSTR + d] = (_Float16)y1;
    }
  }
  __syncthreads();

  // ---------- Phase 3: out_proj (WMMA, 256 -> 128), spill f32 to LDS ----------
#pragma unroll 1
  for (int s = 0; s < 2; ++s) {
    v16h afr[8];                       // hoisted A fragments for this step
#pragma unroll
    for (int kk = 0; kk < 8; ++kk)
      afr[kk] = lds_load_A(xc_lds + (s * 16) * XCSTR + kk * 32, XCSTR, lane);
#pragma unroll 1
    for (int ct = 0; ct < 8; ++ct) {
      v8f acc;
#pragma unroll
      for (int r = 0; r < 8; ++r) acc[r] = 0.f;
#pragma unroll
      for (int kk = 0; kk < 8; ++kk) {
        const v16h b = load_B_f16(Op, DIN, ct * 16, kk * 32, lane);
        acc = wmma16(afr[kk], b, acc);
      }
      const int mbase = (lane >> 4) * 8, n = lane & 15;
#pragma unroll
      for (int r = 0; r < 8; ++r)
        o_lds[(s * 16 + mbase + r) * OSTR + ct * 16 + n] = acc[r];
    }
  }
  __syncthreads();

  // ---------- Phase 4: LayerNorm + SiLU + residual, write output ----------
  {
    const int step = lane >> 4, pos = lane & 15;
    const int p = p0 + pos, b = p >> 14, hw = p & (HWP - 1);
    const float* row = o_lds + (step * 16 + pos) * OSTR;
    float mu = 0.f;
#pragma unroll 4
    for (int c = 0; c < CC; ++c) mu += row[c];
    mu *= (1.f / CC);
    float var = 0.f;
#pragma unroll 4
    for (int c = 0; c < CC; ++c) { const float dv = row[c] - mu; var += dv * dv; }
    var *= (1.f / CC);
    const float rstd = rsqrtf(var + 1e-5f);
    const float* resid = (step == 0 ? x1 : x2) + (size_t)b * CC * HWP + hw;
    float* outp = out + (size_t)step * NELEM + (size_t)b * CC * HWP + hw;
#pragma unroll 2
    for (int c = 0; c < CC; ++c) {
      const float yn = (row[c] - mu) * rstd * ln_w[c] + ln_b[c];
      outp[c * HWP] = resid[c * HWP] + silu_f(yn);
    }
  }
}

extern "C" void kernel_launch(void* const* d_in, const int* in_sizes, int n_in,
                              void* d_out, int out_size, void* d_ws, size_t ws_size,
                              hipStream_t stream) {
  (void)in_sizes; (void)n_in; (void)ws_size; (void)out_size;
  _Float16* wf16 = (_Float16*)d_ws;
  weights_prep_kernel<<<(WS_TOTAL_HALFS + 255) / 256, 256, 0, stream>>>(
      (const float*)d_in[2], (const float*)d_in[5], (const float*)d_in[10], wf16);
  mamba2d_fused_kernel<<<NTOK / 16, 32, 0, stream>>>(
      (const float*)d_in[0], (const float*)d_in[1],
      (const float*)d_in[3], (const float*)d_in[4],
      (const float*)d_in[6], (const float*)d_in[7],
      (const float*)d_in[8], (const float*)d_in[9],
      (const float*)d_in[11], (const float*)d_in[12],
      wf16, (float*)d_out);
}
